// SelfAttention_23261542875519
// MI455X (gfx1250) — matrix operand order repaired and checked by reading, not compiled, not run on hardware
//
#include <hip/hip_runtime.h>

// ---------------------------------------------------------------------------
// Problem constants (from the reference)
// ---------------------------------------------------------------------------
#define NF 8192      // frames
#define MF 1024      // input size
#define DF 1024      // hidden size
#define AP 256       // aperture half width (|i-j| >= AP masked)
#define QSCALE 0.06f

typedef __attribute__((ext_vector_type(16))) __bf16 bf16x16;
typedef __attribute__((ext_vector_type(8)))  __bf16 bf16x8;
typedef __attribute__((ext_vector_type(4)))  __bf16 bf16x4;
typedef __attribute__((ext_vector_type(8)))  float  floatx8;
typedef __attribute__((ext_vector_type(4)))  unsigned int tdm_u32x4;
typedef __attribute__((ext_vector_type(8)))  int          tdm_i32x8;
typedef __attribute__((ext_vector_type(4)))  int          tdm_i32x4;

#ifndef __has_builtin
#define __has_builtin(x) 0
#endif
#if __has_builtin(__builtin_amdgcn_tensor_load_to_lds) && \
    __has_builtin(__builtin_amdgcn_s_wait_tensorcnt)
#define HAVE_TDM 1
#else
#define HAVE_TDM 0
#endif

// LDS panel geometry: K-panel of 32 bf16 per row (64B) + 16B TDM pad -> 80B
// row stride. 80 % 16 == 0 (aligned b128 reads) and row*20 mod 64 spreads 16
// consecutive rows over 16 distinct banks (conflict-free fragment reads).
#define ROWB 80
#define A_PANEL (128 * ROWB)
#define B_PANEL (64 * ROWB)
#define BUF_STRIDE (A_PANEL + B_PANEL)

// ---------------------------------------------------------------------------
// WMMA fragment layouts (CDNA5 16x16x32 bf16, wave32):
// A (16x32): lane l<16 -> row M=l,   K = {k0..k0+7, k0+16..k0+23}
//            lane l>=16 -> row M=l-16, K = {k0+8..15, k0+24..31}
// B (32x16): lane l<16 -> col N=l,   K = {k0..k0+7, k0+16..k0+23}
//            lane l>=16 -> col N=l-16, K = {k0+8..15, k0+24..31}
// C (16x16): VGPR q at lane l -> (M = q + 8*(l>>4), N = l&15)
// ---------------------------------------------------------------------------
__device__ __forceinline__ bf16x16
load_A_frag(const __bf16* A, int lda, int n0, int k0, int lane) {
    const int l = lane & 15, hi = lane >> 4;
    const __bf16* p = A + (size_t)(n0 + l) * lda + k0 + hi * 8;
    bf16x8 lo = *reinterpret_cast<const bf16x8*>(p);
    bf16x8 hh = *reinterpret_cast<const bf16x8*>(p + 16);
    bf16x16 r;
#pragma unroll
    for (int q = 0; q < 8; ++q) { r[q] = lo[q]; r[q + 8] = hh[q]; }
    return r;
}

__device__ __forceinline__ bf16x16
load_Bt_frag_clamp(const __bf16* W, int ldw, int d0, int k0, int lane, int maxRow) {
    const int l = lane & 15, hi = lane >> 4;
    int row = d0 + l;
    row = row < 0 ? 0 : (row >= maxRow ? maxRow - 1 : row);
    const __bf16* p = W + (size_t)row * ldw + k0 + hi * 8;
    bf16x8 a = *reinterpret_cast<const bf16x8*>(p);
    bf16x8 b = *reinterpret_cast<const bf16x8*>(p + 16);
    bf16x16 r;
#pragma unroll
    for (int q = 0; q < 8; ++q) { r[q] = a[q]; r[q + 8] = b[q]; }
    return r;
}

// Fragment loads out of the padded LDS panels (k-local panel, 32 wide).
__device__ __forceinline__ bf16x16
lds_A_frag(const unsigned char* base, int row0, int lane) {
    const int l = lane & 15, hi = lane >> 4;
    const unsigned char* p = base + (row0 + l) * ROWB + hi * 16;
    bf16x8 lo = *reinterpret_cast<const bf16x8*>(p);
    bf16x8 hh = *reinterpret_cast<const bf16x8*>(p + 32);
    bf16x16 r;
#pragma unroll
    for (int q = 0; q < 8; ++q) { r[q] = lo[q]; r[q + 8] = hh[q]; }
    return r;
}

__device__ __forceinline__ bf16x16
lds_B_frag(const unsigned char* base, int row0, int lane) {
    const int l = lane & 15, hi = lane >> 4;
    const unsigned char* p = base + (row0 + l) * ROWB + hi * 16;
    bf16x8 a = *reinterpret_cast<const bf16x8*>(p);
    bf16x8 b = *reinterpret_cast<const bf16x8*>(p + 32);
    bf16x16 r;
#pragma unroll
    for (int q = 0; q < 8; ++q) { r[q] = a[q]; r[q + 8] = b[q]; }
    return r;
}

// A fragment for att.T: A[i][k] = att[jbase+k][i] (f32 column gather -> bf16).
__device__ __forceinline__ bf16x16
load_attT_frag(const float* att, int ldn, int i0, int jbase, int lane, int n) {
    const int l = lane & 15, hi = lane >> 4;
    const int i = i0 + l;
    bf16x16 r;
#pragma unroll
    for (int q = 0; q < 8; ++q) {
        int j0 = jbase + hi * 8 + q;
        int j1 = jbase + 16 + hi * 8 + q;
        float f0 = (j0 >= 0 && j0 < n) ? att[(size_t)j0 * ldn + i] : 0.0f;
        float f1 = (j1 >= 0 && j1 < n) ? att[(size_t)j1 * ldn + i] : 0.0f;
        r[q]     = (__bf16)f0;
        r[q + 8] = (__bf16)f1;
    }
    return r;
}

// B fragment from row-major [K x ncols]: B[k][n] = V[jbase+k][d0+n].
__device__ __forceinline__ bf16x16
load_B_rm_frag(const __bf16* V, int ldv, int jbase, int d0, int lane, int n) {
    const int l = lane & 15, hi = lane >> 4;
    bf16x16 r;
#pragma unroll
    for (int q = 0; q < 16; ++q) {
        int j = jbase + (q < 8 ? hi * 8 + q : 16 + hi * 8 + (q - 8));
        j = j < 0 ? 0 : (j >= n ? n - 1 : j);   // clamped; A side supplies the 0
        r[q] = V[(size_t)j * ldv + d0 + l];
    }
    return r;
}

// ---------------------------------------------------------------------------
// Tensor Data Mover: 2D tile (32 bf16 wide x `rows` tall) -> LDS, with 16B of
// LDS padding every 64B (pad_interval=16 DW, pad_amount=4 DW) => ROWB stride.
// Descriptor packing per CDNA5 ISA ch.8 (D# groups 0/1; groups 2/3 unused).
// ---------------------------------------------------------------------------
#if HAVE_TDM
__device__ __forceinline__ void
tdm_load_tile(void* lds_ptr, const __bf16* gsrc, int rows, int ld_elems) {
    const unsigned long long ga = (unsigned long long)(uintptr_t)gsrc;
    tdm_u32x4 g0;
    g0[0] = 1u;                                        // count=1, user mode
    g0[1] = (unsigned)(uintptr_t)lds_ptr;              // lds_addr (bytes)
    g0[2] = (unsigned)(ga & 0xFFFFFFFFull);            // global_addr[31:0]
    g0[3] = (unsigned)((ga >> 32) & 0x01FFFFFFull)     // global_addr[56:32]
            | 0x80000000u;                             // type=2 ("image")
    tdm_i32x8 g1;
    g1[0] = (int)((1u << 16)      // data_size = 2B
                | (1u << 20)      // pad_enable
                | (3u << 22)      // pad_interval: 16 DWORDs
                | (3u << 25));    // pad_amount:   4 DWORDs
    g1[1] = 0;                                         // td0[15:0]<<16 (td0=1<<24)
    g1[2] = (int)0x00000100u;                          // td0[31:16] | td1[15:0]<<16
    g1[3] = (int)(0x100u | (32u << 16));               // td1[31:16] | tile_dim0=32
    g1[4] = rows;                                      // tile_dim1 (tile_dim2=0)
    g1[5] = ld_elems;                                  // tensor_dim0_stride[31:0]
    g1[6] = 0;
    g1[7] = 0;
    tdm_i32x4 z4 = 0;
#if __clang_major__ >= 23
    tdm_i32x8 z8 = 0;
    __builtin_amdgcn_tensor_load_to_lds(g0, g1, z4, z4, z8, 0);
#else
    __builtin_amdgcn_tensor_load_to_lds(g0, g1, z4, z4, 0);
#endif
}
#endif

// ---------------------------------------------------------------------------
// Kernel 1: f32 -> bf16 conversion
// ---------------------------------------------------------------------------
__global__ void cvt_f32_to_bf16(const float* __restrict__ src,
                                __bf16* __restrict__ dst, long n) {
    long idx = ((long)blockIdx.x * blockDim.x + threadIdx.x) * 4;
    long stride = (long)gridDim.x * blockDim.x * 4;
    for (long i = idx; i < n; i += stride) {
        float4 v = *reinterpret_cast<const float4*>(src + i);
        bf16x4 o;
        o[0] = (__bf16)v.x; o[1] = (__bf16)v.y;
        o[2] = (__bf16)v.z; o[3] = (__bf16)v.w;
        *reinterpret_cast<bf16x4*>(dst + i) = o;
    }
}

// ---------------------------------------------------------------------------
// Kernels 2/6: NT GEMM C[nrows x ncols] = A[nrows x K] * B[ncols x K]^T
// Workgroup = 8 waves computing a 128x64 macro-tile; K-panels of 32 are
// double-buffered in LDS, DMA'd by the TDM (wave 0 issues, TENSORcnt +
// barrier publishes). Each wave owns a 32x32 sub-tile (4 WMMAs / K-step).
// NOTE: buffer pointers are computed arithmetically from `lds` each use —
// initialized arrays of generic pointers into LDS produce addrspacecast
// static initializers that ld.lld rejects.
// ---------------------------------------------------------------------------
template <bool F32OUT>
__global__ __launch_bounds__(256) void
gemm_nt_bf16_lds(const __bf16* __restrict__ A, const __bf16* __restrict__ B,
                 float* __restrict__ Cf, __bf16* __restrict__ Cb,
                 int nrows, int ncols, int kdim) {
    const int lane = threadIdx.x & 31;
    const int wave = threadIdx.x >> 5;   // 0..7
    const int wr = wave >> 1;            // 0..3 : 32-row band
    const int wc = wave & 1;             // 0..1 : 32-col band
    const int bcols = ncols >> 6;
    const int tn = (blockIdx.x / bcols) * 128;
    const int td = (blockIdx.x % bcols) * 64;

    __shared__ __align__(16) unsigned char lds[2 * BUF_STRIDE];

    const int kt = kdim >> 5;
    floatx8 c00 = {}, c01 = {}, c10 = {}, c11 = {};

#if HAVE_TDM
    if (wave == 0) {
        tdm_load_tile(lds,           A + (size_t)tn * kdim, 128, kdim);
        tdm_load_tile(lds + A_PANEL, B + (size_t)td * kdim, 64,  kdim);
    }
#endif
    for (int p = 0; p < kt; ++p) {
#if HAVE_TDM
        const int cur = p & 1;
        unsigned char* curA = lds + cur * BUF_STRIDE;
        unsigned char* curB = curA + A_PANEL;
        if (wave == 0) __builtin_amdgcn_s_wait_tensorcnt((short)0);
        __syncthreads();                       // panel `cur` visible to all
        if (wave == 0 && p + 1 < kt) {         // prefetch next panel (other buf)
            const int nk = (p + 1) * 32;
            unsigned char* nxtA = lds + (cur ^ 1) * BUF_STRIDE;
            tdm_load_tile(nxtA,           A + (size_t)tn * kdim + nk, 128, kdim);
            tdm_load_tile(nxtA + A_PANEL, B + (size_t)td * kdim + nk, 64,  kdim);
        }
#else
        unsigned char* curA = lds;             // synchronous single-buffer path
        unsigned char* curB = lds + A_PANEL;
        const int kb = p * 32;
        __syncthreads();                       // prior reads of panel complete
        {
            const int t = threadIdx.x;
            int row = t >> 1, half = t & 1;    // A: 128 rows x 2 halves
            const __bf16* s = A + (size_t)(tn + row) * kdim + kb + half * 16;
            unsigned char* d = curA + row * ROWB + half * 32;
            *reinterpret_cast<bf16x8*>(d)      = *reinterpret_cast<const bf16x8*>(s);
            *reinterpret_cast<bf16x8*>(d + 16) = *reinterpret_cast<const bf16x8*>(s + 8);
            if (t < 128) {                     // B: 64 rows x 2 halves
                row = t >> 1; half = t & 1;
                s = B + (size_t)(td + row) * kdim + kb + half * 16;
                d = curB + row * ROWB + half * 32;
                *reinterpret_cast<bf16x8*>(d)      = *reinterpret_cast<const bf16x8*>(s);
                *reinterpret_cast<bf16x8*>(d + 16) = *reinterpret_cast<const bf16x8*>(s + 8);
            }
        }
        __syncthreads();
#endif
        bf16x16 a0 = lds_A_frag(curA, wr * 32,      lane);
        bf16x16 a1 = lds_A_frag(curA, wr * 32 + 16, lane);
        bf16x16 b0 = lds_B_frag(curB, wc * 32,      lane);
        bf16x16 b1 = lds_B_frag(curB, wc * 32 + 16, lane);
        c00 = __builtin_amdgcn_wmma_f32_16x16x32_bf16(false, a0, false, b0, (short)0, c00, false, false);
        c01 = __builtin_amdgcn_wmma_f32_16x16x32_bf16(false, a0, false, b1, (short)0, c01, false, false);
        c10 = __builtin_amdgcn_wmma_f32_16x16x32_bf16(false, a1, false, b0, (short)0, c10, false, false);
        c11 = __builtin_amdgcn_wmma_f32_16x16x32_bf16(false, a1, false, b1, (short)0, c11, false, false);
    }

    const int l = lane & 15, hi = lane >> 4;
    const int rn = tn + wr * 32;
    const int cd = td + wc * 32;
#pragma unroll
    for (int q = 0; q < 8; ++q) {
        const int m = q + 8 * hi;
        const size_t r0 = (size_t)(rn + m) * ncols;
        const size_t r1 = (size_t)(rn + 16 + m) * ncols;
        if (F32OUT) {
            Cf[r0 + cd + l]      = c00[q];
            Cf[r0 + cd + 16 + l] = c01[q];
            Cf[r1 + cd + l]      = c10[q];
            Cf[r1 + cd + 16 + l] = c11[q];
        } else {
            Cb[r0 + cd + l]      = (__bf16)c00[q];
            Cb[r0 + cd + 16 + l] = (__bf16)c01[q];
            Cb[r1 + cd + l]      = (__bf16)c10[q];
            Cb[r1 + cd + 16 + l] = (__bf16)c11[q];
        }
    }
}

// ---------------------------------------------------------------------------
// Kernel 3: banded raw logits. One wave per (16-row strip, 16-col tile);
// 33 col tiles cover the band. Raw f32 logits -> att region of d_out.
// ---------------------------------------------------------------------------
__global__ void logits_band_kernel(const __bf16* __restrict__ Qh,
                                   const __bf16* __restrict__ Kh,
                                   float* __restrict__ att, int n, int dd) {
    const int lane = threadIdx.x & 31;
    const int it = blockIdx.x / 33;
    const int jt = blockIdx.x % 33;
    const int i0 = it * 16;
    const int j0 = i0 - AP + jt * 16;
    if (j0 + 15 < 0 || j0 >= n) return;
    floatx8 c = {};
    for (int k = 0; k < dd; k += 32) {
        bf16x16 a = load_A_frag(Qh, dd, i0, k, lane);
        bf16x16 b = load_Bt_frag_clamp(Kh, dd, j0, k, lane, n);
        c = __builtin_amdgcn_wmma_f32_16x16x32_bf16(false, a, false, b, (short)0, c, false, false);
    }
    const int l = lane & 15, hi = lane >> 4;
    const int col = j0 + l;
    if (col >= 0 && col < n) {
#pragma unroll
        for (int q = 0; q < 8; ++q) {
            const int row = i0 + q + 8 * hi;
            att[(size_t)row * n + col] = c[q];
        }
    }
}

// ---------------------------------------------------------------------------
// Kernel 4: banded row softmax (applies QSCALE, masks diagonal), writes the
// FULL row (zeros outside the band -> also initializes the att output).
// ---------------------------------------------------------------------------
__global__ void softmax_band_kernel(float* __restrict__ att, int n) {
    const int i = blockIdx.x;
    const int t = threadIdx.x;
    __shared__ float red[256];
    int jb0 = i - (AP - 1); if (jb0 < 0) jb0 = 0;
    int jb1 = i + (AP - 1); if (jb1 > n - 1) jb1 = n - 1;
    float* row = att + (size_t)i * n;

    float mx = -3.0e38f;
    for (int j = jb0 + t; j <= jb1; j += 256)
        if (j != i) mx = fmaxf(mx, row[j] * QSCALE);
    red[t] = mx; __syncthreads();
    for (int s = 128; s > 0; s >>= 1) {
        if (t < s) red[t] = fmaxf(red[t], red[t + s]);
        __syncthreads();
    }
    mx = red[0]; __syncthreads();

    float sum = 0.0f;
    for (int j = jb0 + t; j <= jb1; j += 256)
        if (j != i) sum += __expf(row[j] * QSCALE - mx);
    red[t] = sum; __syncthreads();
    for (int s = 128; s > 0; s >>= 1) {
        if (t < s) red[t] += red[t + s];
        __syncthreads();
    }
    const float inv = 1.0f / red[0];

    for (int c = t; c < n; c += 256) {
        float v = 0.0f;
        if (c >= jb0 && c <= jb1 && c != i)
            v = __expf(row[c] * QSCALE - mx) * inv;
        row[c] = v;
    }
}

// ---------------------------------------------------------------------------
// Kernel 5: banded transposed einsum y1[i,d] = sum_j att[j,i] * V[j,d]
// ---------------------------------------------------------------------------
__global__ void einsum_band_kernel(const float* __restrict__ att,
                                   const __bf16* __restrict__ Vh,
                                   __bf16* __restrict__ Y1h, int n, int dd) {
    const int lane = threadIdx.x & 31;
    const int dtiles = dd >> 4;
    const int i0 = (blockIdx.x / dtiles) * 16;
    const int d0 = (blockIdx.x % dtiles) * 16;
    floatx8 c = {};
    for (int tch = 0; tch < 17; ++tch) {
        const int jbase = i0 - AP + tch * 32;
        if (jbase + 31 < 0 || jbase >= n) continue;
        bf16x16 a = load_attT_frag(att, n, i0, jbase, lane, n);
        bf16x16 b = load_B_rm_frag(Vh, dd, jbase, d0, lane, n);
        c = __builtin_amdgcn_wmma_f32_16x16x32_bf16(false, a, false, b, (short)0, c, false, false);
    }
    const int l = lane & 15, hi = lane >> 4;
#pragma unroll
    for (int q = 0; q < 8; ++q) {
        const int m = q + 8 * hi;
        Y1h[(size_t)(i0 + m) * dd + d0 + l] = (__bf16)c[q];
    }
}

// ---------------------------------------------------------------------------
// Host-side launcher
// ---------------------------------------------------------------------------
extern "C" void kernel_launch(void* const* d_in, const int* in_sizes, int n_in,
                              void* d_out, int out_size, void* d_ws, size_t ws_size,
                              hipStream_t stream) {
    const float* x  = (const float*)d_in[0];
    const float* Wk = (const float*)d_in[1];
    const float* Wq = (const float*)d_in[2];
    const float* Wv = (const float*)d_in[3];
    const float* Wo = (const float*)d_in[4];

    float* y_out = (float*)d_out;                          // [NF x MF]
    float* att   = y_out + (size_t)NF * MF;                // [NF x NF]

    const size_t XE = (size_t)NF * MF;                     // 8M elems
    const size_t WE = (size_t)DF * MF;                     // 1M elems
    __bf16* xh  = (__bf16*)d_ws;
    __bf16* Wkh = xh  + XE;
    __bf16* Wqh = Wkh + WE;
    __bf16* Wvh = Wqh + WE;
    __bf16* Woh = Wvh + WE;
    __bf16* Kh  = Woh + WE;
    __bf16* Qh  = Kh  + XE;
    __bf16* Vh  = Qh  + XE;
    __bf16* Y1h = Vh  + XE;
    (void)ws_size; (void)in_sizes; (void)n_in; (void)out_size;

    // 1) f32 -> bf16 conversions
    cvt_f32_to_bf16<<<2048, 256, 0, stream>>>(x,  xh,  (long)XE);
    cvt_f32_to_bf16<<<512,  256, 0, stream>>>(Wk, Wkh, (long)WE);
    cvt_f32_to_bf16<<<512,  256, 0, stream>>>(Wq, Wqh, (long)WE);
    cvt_f32_to_bf16<<<512,  256, 0, stream>>>(Wv, Wvh, (long)WE);
    cvt_f32_to_bf16<<<512,  256, 0, stream>>>(Wo, Woh, (long)WE);

    // 2) K/Q/V projections (TDM-staged, LDS double-buffered WMMA GEMM)
    const int gproj = (NF / 128) * (DF / 64);              // 1024 blocks
    gemm_nt_bf16_lds<false><<<gproj, 256, 0, stream>>>(xh, Wkh, nullptr, Kh, NF, DF, MF);
    gemm_nt_bf16_lds<false><<<gproj, 256, 0, stream>>>(xh, Wqh, nullptr, Qh, NF, DF, MF);
    gemm_nt_bf16_lds<false><<<gproj, 256, 0, stream>>>(xh, Wvh, nullptr, Vh, NF, DF, MF);

    // 3) banded raw logits into att region of d_out
    logits_band_kernel<<<(NF / 16) * 33, 32, 0, stream>>>(Qh, Kh, att, NF, DF);

    // 4) banded softmax (+ zero fill of the full att matrix)
    softmax_band_kernel<<<NF, 256, 0, stream>>>(att, NF);

    // 5) y1 = att.T @ V (banded), bf16 out
    einsum_band_kernel<<<(NF / 16) * (DF / 16), 32, 0, stream>>>(att, Vh, Y1h, NF, DF);

    // 6) y = y1 @ Wo.T, f32 out into d_out
    const int gout = (NF / 128) * (MF / 64);
    gemm_nt_bf16_lds<true><<<gout, 256, 0, stream>>>(Y1h, Woh, y_out, nullptr, NF, MF, DF);
}
